// EncoderBlock_18597208392159
// MI455X (gfx1250) — matrix operand order repaired and checked by reading, not compiled, not run on hardware
//
#include <hip/hip_runtime.h>
#include <hip/hip_bf16.h>
#include <math.h>
#include <stdint.h>

typedef __bf16 bf16;
typedef __attribute__((ext_vector_type(16))) __bf16 v16bf;
typedef __attribute__((ext_vector_type(8)))  float  v8f;
typedef __attribute__((ext_vector_type(4)))  float  f4;
typedef int v4i __attribute__((vector_size(16)));

#define D_MODEL 1024
#define NHEAD   16
#define DK      64
#define DFF     4096
#define BATCH   4
#define SEQ     2048
#define NTOK    (BATCH*SEQ)

union AFrag { v16bf v; f4 f[2]; };

// CDNA5 async global->LDS path (ASYNCcnt), guarded so compile never breaks.
#if defined(__HIP_DEVICE_COMPILE__) && \
    __has_builtin(__builtin_amdgcn_global_load_async_to_lds_b128) && \
    __has_builtin(__builtin_amdgcn_s_wait_asynccnt)
#define USE_ASYNC_LDS 1
#define AS1P(p) ((__attribute__((address_space(1))) v4i*)(uintptr_t)(p))
#define AS3P(p) ((__attribute__((address_space(3))) v4i*)(unsigned)(uintptr_t)(p))
#else
#define USE_ASYNC_LDS 0
#endif

__device__ __forceinline__ bf16 f2b(float x) {
    // round-to-nearest-even fp32 -> bf16, no reliance on __bf16 arithmetic
    unsigned u = __builtin_bit_cast(unsigned, x);
    unsigned r = (u + 0x7fffu + ((u >> 16) & 1u)) >> 16;
    return __builtin_bit_cast(bf16, (unsigned short)r);
}

// ---------------------------------------------------------------------------
// fp32 [K,N] -> bf16 transposed [N,K]
// ---------------------------------------------------------------------------
__global__ void wxpose(const float* __restrict__ in, bf16* __restrict__ out,
                       int K, int N) {
    size_t idx = (size_t)blockIdx.x * blockDim.x + threadIdx.x;
    if (idx >= (size_t)K * N) return;
    int k = (int)(idx / N), n = (int)(idx % N);
    out[(size_t)n * K + k] = f2b(in[idx]);
}

// ---------------------------------------------------------------------------
// bf16 V [B*S, D] -> Vt [B,H,DK,S]  (so PV B-frags load contiguously)
// ---------------------------------------------------------------------------
__global__ void vxpose(const bf16* __restrict__ v, bf16* __restrict__ vt) {
    size_t idx = (size_t)blockIdx.x * blockDim.x + threadIdx.x;
    int d = (int)(idx % D_MODEL);
    int t = (int)(idx / D_MODEL);
    int b = t / SEQ, s = t % SEQ;
    int h = d / DK, dd = d % DK;
    vt[(((size_t)(b * NHEAD + h)) * DK + dd) * SEQ + s] = v[idx];
}

// ---------------------------------------------------------------------------
// LayerNorm (ddof=1 std, eps added to std) fp32 -> bf16
// ---------------------------------------------------------------------------
__global__ __launch_bounds__(256) void ln_bf16(
    const float* __restrict__ x, const float* __restrict__ alpha,
    const float* __restrict__ beta, bf16* __restrict__ out) {
    __shared__ float s1[256], s2[256];
    const int row = blockIdx.x, tid = threadIdx.x;
    const float* xr = x + (size_t)row * D_MODEL;
    float v[4], s = 0.f, q = 0.f;
#pragma unroll
    for (int i = 0; i < 4; ++i) { v[i] = xr[tid + i * 256]; s += v[i]; q += v[i] * v[i]; }
    s1[tid] = s; s2[tid] = q;
    __syncthreads();
    for (int off = 128; off > 0; off >>= 1) {
        if (tid < off) { s1[tid] += s1[tid + off]; s2[tid] += s2[tid + off]; }
        __syncthreads();
    }
    float mean = s1[0] / (float)D_MODEL;
    float var  = (s2[0] - (float)D_MODEL * mean * mean) / (float)(D_MODEL - 1);
    float inv  = 1.0f / (sqrtf(var > 0.f ? var : 0.f) + 1e-6f);
    bf16* orow = out + (size_t)row * D_MODEL;
#pragma unroll
    for (int i = 0; i < 4; ++i) {
        int c = tid + i * 256;
        orow[c] = f2b(alpha[c] * (v[i] - mean) * inv + beta[c]);
    }
}

// ---------------------------------------------------------------------------
// bf16 GEMM: C[M,N] = A[M,K] * BT[N,K]^T (+bias)(+resid)(relu)
// 256 threads = 8 waves; block tile 128x128; triple-buffered LDS with a
// depth-2 async-copy pipeline (s_wait_asynccnt <= 4 keeps one stage in
// flight across the WMMA block); per wave 32x64 = 8 WMMAs per k-step.
// ---------------------------------------------------------------------------
__device__ __forceinline__ void stage_tile(const bf16* __restrict__ gA,
                                           const bf16* __restrict__ gB,
                                           bf16* lA, bf16* lB, int tid, int K) {
    // 128 rows x 32 cols bf16 per tile = 512 16B chunks; 2 chunks/thread each
#pragma unroll
    for (int j = 0; j < 2; ++j) {
        const int c = tid * 2 + j;
        const int row = c >> 2, q = (c & 3) * 8;
        const bf16* ga = gA + (size_t)row * K + q;
        const bf16* gb = gB + (size_t)row * K + q;
        bf16* la = lA + row * 32 + q;
        bf16* lb = lB + row * 32 + q;
#if USE_ASYNC_LDS
        __builtin_amdgcn_global_load_async_to_lds_b128(AS1P(ga), AS3P(la), 0, 0);
        __builtin_amdgcn_global_load_async_to_lds_b128(AS1P(gb), AS3P(lb), 0, 0);
#else
        *(f4*)la = *(const f4*)ga;
        *(f4*)lb = *(const f4*)gb;
#endif
    }
}

__global__ __launch_bounds__(256) void gemm_bf16(
    const bf16* __restrict__ A, const bf16* __restrict__ BT,
    int M, int N, int K,
    const float* __restrict__ bias, const float* __restrict__ resid,
    float* __restrict__ outF, bf16* __restrict__ outB, int relu) {
    __shared__ __align__(16) bf16 As[3][128 * 32];
    __shared__ __align__(16) bf16 Bs[3][128 * 32];

    const int tid = threadIdx.x;
    const int wid = tid >> 5, lane = tid & 31;
    const int half = lane >> 4, l16 = lane & 15;
    const int mBase = blockIdx.y * 128, nBase = blockIdx.x * 128;
    const int wm = (wid & 3) * 32;       // wave M offset in block tile
    const int wn = (wid >> 2) * 64;      // wave N offset in block tile

    v8f acc[2][4];
#pragma unroll
    for (int im = 0; im < 2; ++im)
#pragma unroll
        for (int in = 0; in < 4; ++in) acc[im][in] = {};

    const bf16* gA = A + (size_t)mBase * K;
    const bf16* gB = BT + (size_t)nBase * K;
    const int ksteps = K >> 5;

    // depth-2 pipeline prologue
    stage_tile(gA, gB, As[0], Bs[0], tid, K);
    if (1 < ksteps)
        stage_tile(gA + 32, gB + 32, As[1], Bs[1], tid, K);

    int buf = 0;
    for (int i = 0; i < ksteps; ++i) {
        // stage(i) resident; stage(i+1) may remain in flight (in-order cnt)
#if USE_ASYNC_LDS
        if (i + 1 < ksteps) __builtin_amdgcn_s_wait_asynccnt(4);
        else                __builtin_amdgcn_s_wait_asynccnt(0);
#endif
        __syncthreads();

        const bf16* a_s = As[buf];
        const bf16* b_s = Bs[buf];

        AFrag a[2], b[4];
#pragma unroll
        for (int im = 0; im < 2; ++im) {
            const bf16* ap = a_s + (wm + im * 16 + l16) * 32;
            a[im].f[0] = *(const f4*)(ap + 8 * half);
            a[im].f[1] = *(const f4*)(ap + 16 + 8 * half);
        }
#pragma unroll
        for (int in = 0; in < 4; ++in) {
            const bf16* bp = b_s + (wn + in * 16 + l16) * 32 + 8 * half;
            b[in].f[0] = *(const f4*)(bp);
            b[in].f[1] = *(const f4*)(bp + 16);
        }

        if (i + 2 < ksteps) {            // issue stage(i+2) into retired buffer
            int nb = buf + 2; if (nb >= 3) nb -= 3;
            stage_tile(gA + (i + 2) * 32, gB + (i + 2) * 32,
                       As[nb], Bs[nb], tid, K);
        }

#pragma unroll
        for (int im = 0; im < 2; ++im)
#pragma unroll
            for (int in = 0; in < 4; ++in)
                acc[im][in] = __builtin_amdgcn_wmma_f32_16x16x32_bf16(
                    false, a[im].v, false, b[in].v, (short)0, acc[im][in], false, false);

        if (++buf >= 3) buf = 0;
    }

#pragma unroll
    for (int in = 0; in < 4; ++in) {
        const int col = nBase + wn + in * 16 + l16;
        const float bv = bias ? bias[col] : 0.0f;
#pragma unroll
        for (int im = 0; im < 2; ++im) {
#pragma unroll
            for (int r = 0; r < 8; ++r) {
                const int row = mBase + wm + im * 16 + r + 8 * half;
                const size_t idx = (size_t)row * N + col;
                float v = acc[im][in][r] + bv;
                if (resid) v += resid[idx];
                if (relu)  v = v > 0.f ? v : 0.f;
                if (outF)  outF[idx] = v;
                if (outB)  outB[idx] = f2b(v);
            }
        }
    }
}

// ---------------------------------------------------------------------------
// Flash-style attention: one wave per 16-row Q tile, keys streamed 32 at a time
// ---------------------------------------------------------------------------
__global__ __launch_bounds__(256) void attn_kernel(
    const bf16* __restrict__ q, const bf16* __restrict__ k,
    const bf16* __restrict__ vt, const int* __restrict__ mask,
    bf16* __restrict__ out) {
    __shared__ __align__(16) bf16 pbuf[8][16 * 32];  // wave-private P tiles
    const int tid = threadIdx.x;
    const int wid = tid >> 5, lane = tid & 31;
    const int half = lane >> 4, l16 = lane & 15;
    const int bh = blockIdx.x, b = bh / NHEAD, h = bh % NHEAD;
    const int q0 = blockIdx.y * 128 + wid * 16;

    // Q A-frags: dk=64 -> two 16x32 k-steps, invariant over key loop
    AFrag qa[2];
    {
        const bf16* qp = q + ((size_t)(b * SEQ + q0 + l16)) * D_MODEL + h * DK;
#pragma unroll
        for (int kk = 0; kk < 2; ++kk) {
            qa[kk].f[0] = *(const f4*)(qp + kk * 32 + 8 * half);
            qa[kk].f[1] = *(const f4*)(qp + kk * 32 + 16 + 8 * half);
        }
    }

    v8f o[4]; o[0] = {}; o[1] = {}; o[2] = {}; o[3] = {};
    float m[8], l[8];
#pragma unroll
    for (int r = 0; r < 8; ++r) { m[r] = -1e30f; l[r] = 0.f; }

    bf16* pls = &pbuf[wid][0];

    for (int key0 = 0; key0 < SEQ; key0 += 32) {
        v8f c[2];
#pragma unroll
        for (int t = 0; t < 2; ++t) {
            const int key = key0 + t * 16 + l16;
            const bf16* kp = k + ((size_t)(b * SEQ + key)) * D_MODEL + h * DK + 8 * half;
            AFrag kb0, kb1;
            kb0.f[0] = *(const f4*)(kp);       kb0.f[1] = *(const f4*)(kp + 16);
            kb1.f[0] = *(const f4*)(kp + 32);  kb1.f[1] = *(const f4*)(kp + 48);
            v8f cc = {};
            cc = __builtin_amdgcn_wmma_f32_16x16x32_bf16(false, qa[0].v, false, kb0.v, (short)0, cc, false, false);
            cc = __builtin_amdgcn_wmma_f32_16x16x32_bf16(false, qa[1].v, false, kb1.v, (short)0, cc, false, false);
            const float madd = (mask[b * SEQ + key] == 0) ? -1e9f : 0.0f;
#pragma unroll
            for (int r = 0; r < 8; ++r) cc[r] = cc[r] * 0.125f + madd;
            c[t] = cc;
        }
        // online softmax (rows live in the 16-lane half; xor masks stay inside it)
        float fac[8];
#pragma unroll
        for (int r = 0; r < 8; ++r) {
            float mx = fmaxf(c[0][r], c[1][r]);
            mx = fmaxf(mx, __shfl_xor(mx, 1, 32));
            mx = fmaxf(mx, __shfl_xor(mx, 2, 32));
            mx = fmaxf(mx, __shfl_xor(mx, 4, 32));
            mx = fmaxf(mx, __shfl_xor(mx, 8, 32));
            const float nm = fmaxf(m[r], mx);
            fac[r] = __expf(m[r] - nm);
            m[r] = nm;
            const float p0 = __expf(c[0][r] - nm);
            const float p1 = __expf(c[1][r] - nm);
            c[0][r] = p0; c[1][r] = p1;
            float rs = p0 + p1;
            rs += __shfl_xor(rs, 1, 32);
            rs += __shfl_xor(rs, 2, 32);
            rs += __shfl_xor(rs, 4, 32);
            rs += __shfl_xor(rs, 8, 32);
            l[r] = l[r] * fac[r] + rs;
        }
#pragma unroll
        for (int t2 = 0; t2 < 4; ++t2)
#pragma unroll
            for (int r = 0; r < 8; ++r) o[t2][r] *= fac[r];

        // repack P (C-layout) -> A-frag layout via wave-private LDS tile
#pragma unroll
        for (int r = 0; r < 8; ++r) {
            const int R = r + 8 * half;
            pls[R * 32 + l16]      = f2b(c[0][r]);
            pls[R * 32 + 16 + l16] = f2b(c[1][r]);
        }
        AFrag pa;
        {
            const bf16* pp = pls + l16 * 32;
            pa.f[0] = *(const f4*)(pp + 8 * half);
            pa.f[1] = *(const f4*)(pp + 16 + 8 * half);
        }
        // O += P @ V  (Vt rows are contiguous over keys)
#pragma unroll
        for (int t2 = 0; t2 < 4; ++t2) {
            const bf16* vp = vt + ((size_t)(bh * DK + t2 * 16 + l16)) * SEQ + key0 + 8 * half;
            AFrag vb;
            vb.f[0] = *(const f4*)(vp);
            vb.f[1] = *(const f4*)(vp + 16);
            o[t2] = __builtin_amdgcn_wmma_f32_16x16x32_bf16(false, pa.v, false, vb.v, (short)0, o[t2], false, false);
        }
    }

    float il[8];
#pragma unroll
    for (int r = 0; r < 8; ++r) il[r] = 1.0f / l[r];
#pragma unroll
    for (int t2 = 0; t2 < 4; ++t2)
#pragma unroll
        for (int r = 0; r < 8; ++r) {
            const int srow = q0 + r + 8 * half;
            const int col = h * DK + t2 * 16 + l16;
            out[((size_t)(b * SEQ + srow)) * D_MODEL + col] = f2b(o[t2][r] * il[r]);
        }
}

// ---------------------------------------------------------------------------
extern "C" void kernel_launch(void* const* d_in, const int* in_sizes, int n_in,
                              void* d_out, int out_size, void* d_ws, size_t ws_size,
                              hipStream_t stream) {
    (void)in_sizes; (void)n_in; (void)out_size; (void)ws_size;
    const float* x    = (const float*)d_in[0];
    const int*   mask = (const int*)d_in[1];
    const float* wq   = (const float*)d_in[2];
    const float* wk   = (const float*)d_in[3];
    const float* wv   = (const float*)d_in[4];
    const float* wo   = (const float*)d_in[5];
    const float* w1   = (const float*)d_in[6];
    const float* b1   = (const float*)d_in[7];
    const float* w2   = (const float*)d_in[8];
    const float* b2   = (const float*)d_in[9];
    const float* ln1a = (const float*)d_in[10];
    const float* ln1b = (const float*)d_in[11];
    const float* ln2a = (const float*)d_in[12];
    const float* ln2b = (const float*)d_in[13];

    char* ws = (char*)d_ws;
    const size_t MB = 1024 * 1024;
    bf16*  wqT  = (bf16*)(ws + 0 * MB);     // [1024,1024] 2MB
    bf16*  wkT  = (bf16*)(ws + 2 * MB);
    bf16*  wvT  = (bf16*)(ws + 4 * MB);
    bf16*  woT  = (bf16*)(ws + 6 * MB);
    bf16*  w1T  = (bf16*)(ws + 8 * MB);     // [4096,1024] 8MB
    bf16*  w2T  = (bf16*)(ws + 16 * MB);    // [1024,4096] 8MB
    bf16*  xn   = (bf16*)(ws + 24 * MB);    // [8192,1024] 16MB, reused as attn out
    bf16*  qb   = (bf16*)(ws + 40 * MB);    // 16MB, reused as xn2
    bf16*  kb   = (bf16*)(ws + 56 * MB);    // 16MB
    bf16*  vb   = (bf16*)(ws + 72 * MB);    // 16MB
    bf16*  vt   = (bf16*)(ws + 88 * MB);    // 16MB
    bf16*  hb   = (bf16*)(ws + 56 * MB);    // [8192,4096] 64MB overlays k/v/vt
    float* x1   = (float*)(ws + 120 * MB);  // fp32 residual 32MB
    bf16*  attn = xn;
    bf16*  xn2  = qb;

    dim3 blk(256);
    {   // weight prep: fp32 -> bf16 transposed
        int g1 = (1024 * 1024 + 255) / 256;
        wxpose<<<g1, blk, 0, stream>>>(wq, wqT, 1024, 1024);
        wxpose<<<g1, blk, 0, stream>>>(wk, wkT, 1024, 1024);
        wxpose<<<g1, blk, 0, stream>>>(wv, wvT, 1024, 1024);
        wxpose<<<g1, blk, 0, stream>>>(wo, woT, 1024, 1024);
        int g4 = (1024 * 4096 + 255) / 256;
        wxpose<<<g4, blk, 0, stream>>>(w1, w1T, 1024, 4096);
        wxpose<<<g4, blk, 0, stream>>>(w2, w2T, 4096, 1024);
    }
    ln_bf16<<<NTOK, blk, 0, stream>>>(x, ln1a, ln1b, xn);
    gemm_bf16<<<dim3(8, 64), blk, 0, stream>>>(xn, wqT, NTOK, 1024, 1024, nullptr, nullptr, nullptr, qb, 0);
    gemm_bf16<<<dim3(8, 64), blk, 0, stream>>>(xn, wkT, NTOK, 1024, 1024, nullptr, nullptr, nullptr, kb, 0);
    gemm_bf16<<<dim3(8, 64), blk, 0, stream>>>(xn, wvT, NTOK, 1024, 1024, nullptr, nullptr, nullptr, vb, 0);
    vxpose<<<(NTOK * D_MODEL) / 256, blk, 0, stream>>>(vb, vt);
    attn_kernel<<<dim3(BATCH * NHEAD, SEQ / 128), blk, 0, stream>>>(qb, kb, vt, mask, attn);
    gemm_bf16<<<dim3(8, 64), blk, 0, stream>>>(attn, woT, NTOK, 1024, 1024, nullptr, x, x1, nullptr, 0);
    ln_bf16<<<NTOK, blk, 0, stream>>>(x1, ln2a, ln2b, xn2);
    gemm_bf16<<<dim3(32, 64), blk, 0, stream>>>(xn2, w1T, NTOK, 4096, 1024, b1, nullptr, nullptr, hb, 1);
    gemm_bf16<<<dim3(8, 64), blk, 0, stream>>>(hb, w2T, NTOK, 1024, 4096, b2, x1, (float*)d_out, nullptr, 0);
}
